// PointNet_FP_Module_64381559767677
// MI455X (gfx1250) — matrix operand order repaired and checked by reading, not compiled, not run on hardware
//
#include <hip/hip_runtime.h>
#include <hip/hip_bf16.h>

// ---------------------------------------------------------------------------
// Types for CDNA5 WMMA (wave32): v_wmma_f32_16x16x32_bf16
// ---------------------------------------------------------------------------
typedef __attribute__((ext_vector_type(16))) __bf16 v16bf;
typedef __attribute__((ext_vector_type(8)))  float  v8f;
typedef int v4i_gcc __attribute__((vector_size(16)));   // b128 payload type of the async builtin

union Frag16 {
    unsigned int u[8];   // 8 dwords = 16 packed bf16
    v16bf        v;
};

// Problem constants (from setup_inputs)
#define BATCH 8
#define NPTS  8192      // N
#define MPTS  2048      // M
#define C1CH  128
#define C2CH  256
#define CIN   384       // C2 + C1
#define HCH   256       // H1 == H2

#define LDS_PITCH 80    // bytes per staged X row (16B aligned, 20 dwords -> conflict-free)

__device__ __forceinline__ unsigned short f32_to_bf16(float f) {
    unsigned int u = __float_as_uint(f);
    unsigned int r = u + 0x7FFFu + ((u >> 16) & 1u);   // round-to-nearest-even
    return (unsigned short)(r >> 16);
}

// ---------------------------------------------------------------------------
// Async global->LDS 16-byte copy (CDNA5, ASYNCcnt-tracked)
// ---------------------------------------------------------------------------
__device__ __forceinline__ void async_copy_b128(void* lds_dst, const void* gsrc) {
#if __has_builtin(__builtin_amdgcn_global_load_async_to_lds_b128)
    __builtin_amdgcn_global_load_async_to_lds_b128(
        (__attribute__((address_space(1))) v4i_gcc*)gsrc,
        (__attribute__((address_space(3))) v4i_gcc*)lds_dst, 0, 0);
#else
    auto l3 = (__attribute__((address_space(3))) unsigned char*)lds_dst;
    auto g1 = (const __attribute__((address_space(1))) unsigned char*)gsrc;
    asm volatile("global_load_async_to_lds_b128 %0, %1, off"
                 :: "v"(l3), "v"(g1) : "memory");
#endif
}

__device__ __forceinline__ void wait_async_le1() {
#if __has_builtin(__builtin_amdgcn_s_wait_asynccnt)
    __builtin_amdgcn_s_wait_asynccnt(1);
#else
    asm volatile("s_wait_asynccnt 0x1" ::: "memory");
#endif
}
__device__ __forceinline__ void wait_async_le0() {
#if __has_builtin(__builtin_amdgcn_s_wait_asynccnt)
    __builtin_amdgcn_s_wait_asynccnt(0);
#else
    asm volatile("s_wait_asynccnt 0x0" ::: "memory");
#endif
}

// ---------------------------------------------------------------------------
// Kernel 1: convert W1 (256x384) and W2 (256x256) f32 -> bf16
// ---------------------------------------------------------------------------
__global__ void cvt_weights(const float* __restrict__ W1, const float* __restrict__ W2,
                            unsigned short* __restrict__ W1b, unsigned short* __restrict__ W2b) {
    int i = blockIdx.x * blockDim.x + threadIdx.x;
    if (i < HCH * CIN) W1b[i] = f32_to_bf16(W1[i]);
    if (i < HCH * HCH) W2b[i] = f32_to_bf16(W2[i]);
}

// ---------------------------------------------------------------------------
// Kernel 2: 3-NN + inverse-distance interpolation + concat -> X (B*N, 384) bf16
// One block = 256 query points of one batch; xyz2 tile broadcast via LDS.
// ---------------------------------------------------------------------------
__global__ void knn_interp(const float* __restrict__ xyz1, const float* __restrict__ xyz2,
                           const float* __restrict__ points1, const float* __restrict__ points2,
                           unsigned short* __restrict__ Xb) {
    __shared__ float sx[MPTS], sy[MPTS], sz[MPTS];
    const int blocksPerBatch = NPTS / 256;
    int b    = blockIdx.x / blocksPerBatch;
    int nblk = blockIdx.x % blocksPerBatch;
    int tid  = threadIdx.x;

    const float* x2 = xyz2 + (size_t)b * 3 * MPTS;
    for (int m = tid; m < MPTS; m += 256) {
        sx[m] = x2[m];
        sy[m] = x2[MPTS + m];
        sz[m] = x2[2 * MPTS + m];
    }
    __syncthreads();

    int n = nblk * 256 + tid;
    const float* x1 = xyz1 + (size_t)b * 3 * NPTS;
    float px = x1[n], py = x1[NPTS + n], pz = x1[2 * NPTS + n];

    float d0 = 3.4e38f, d1 = 3.4e38f, d2 = 3.4e38f;
    int   i0 = 0, i1 = 0, i2 = 0;
    for (int m = 0; m < MPTS; ++m) {
        float dx = px - sx[m], dy = py - sy[m], dz = pz - sz[m];
        float d  = dx * dx + dy * dy + dz * dz;
        if (d < d0)      { d2 = d1; i2 = i1; d1 = d0; i1 = i0; d0 = d; i0 = m; }
        else if (d < d1) { d2 = d1; i2 = i1; d1 = d;  i1 = m; }
        else if (d < d2) { d2 = d;  i2 = m; }
    }
    d0 = fmaxf(d0, 1e-10f); d1 = fmaxf(d1, 1e-10f); d2 = fmaxf(d2, 1e-10f);
    float r0 = 1.f / (d0 + 1e-8f), r1 = 1.f / (d1 + 1e-8f), r2 = 1.f / (d2 + 1e-8f);
    float rs = 1.f / (r0 + r1 + r2);
    float w0 = r0 * rs, w1 = r1 * rs, w2 = r2 * rs;

    const float* p2 = points2 + (size_t)b * C2CH * MPTS;
    unsigned short* xrow = Xb + ((size_t)b * NPTS + n) * CIN;
    for (int c = 0; c < C2CH; ++c) {
        const float* pc = p2 + (size_t)c * MPTS;
        float v = w0 * pc[i0] + w1 * pc[i1] + w2 * pc[i2];
        xrow[c] = f32_to_bf16(v);
    }
    const float* p1 = points1 + (size_t)b * C1CH * NPTS;
    for (int c = 0; c < C1CH; ++c)
        xrow[C2CH + c] = f32_to_bf16(p1[(size_t)c * NPTS + n]);
}

// ---------------------------------------------------------------------------
// Kernel 3: WMMA GEMM  Y(B,H,N) = W(H,K)bf16 x X(B*N,K)bf16, f32 accumulate.
// Block = 8 waves = 128 output rows x 64 cols.  Per k-step the shared 64x32
// X tile (4KB) is staged to LDS with async global->LDS b128 copies (double
// buffered, ASYNCcnt + barrier), then all 8 waves consume it as B-fragments.
// One wave: 1 A-fragment (global, L2-resident W) + 4 B-fragments (LDS) ->
// 4x v_wmma_f32_16x16x32_bf16.
// ---------------------------------------------------------------------------
__global__ void gemm_wmma_bf16(const unsigned short* __restrict__ Wb,
                               const unsigned short* __restrict__ Xb,
                               float* __restrict__ Y,
                               int K) {
    __shared__ __align__(16) unsigned char lds[2][64 * LDS_PITCH];  // 2 x 5KB

    const int colTiles  = NPTS / 64;   // 128
    const int rowBlocks = HCH / 128;   // 2
    int blk = blockIdx.x;
    int b   = blk / (rowBlocks * colTiles);
    int rm  = blk % (rowBlocks * colTiles);
    int rb  = rm / colTiles;
    int ct  = rm % colTiles;
    int tid  = threadIdx.x;
    int wave = tid >> 5;
    int lane = tid & 31;
    int lm = lane & 15;   // row-in-tile (A) / col-in-tile (B, C/D)
    int hf = lane >> 4;   // lane half
    int rt = rb * 8 + wave;

    // async staging: thread -> one 16B chunk of the 64x32 bf16 tile
    int crow = tid >> 2, cchunk = tid & 3;
    const unsigned char* gsrc = (const unsigned char*)Xb
        + ((size_t)b * NPTS + (size_t)ct * 64 + crow) * K * 2 + cchunk * 16;
    unsigned int loff = (unsigned int)crow * LDS_PITCH + cchunk * 16;

    const unsigned int* Wp = (const unsigned int*)Wb;
    size_t arow = ((size_t)(rt * 16 + lm) * K) >> 1;   // dword index of A row

    v8f acc0 = {}, acc1 = {}, acc2 = {}, acc3 = {};

    const int steps = K >> 5;
    async_copy_b128(&lds[0][loff], gsrc);              // prologue: k-step 0
    for (int i = 0; i < steps; ++i) {
        if (i + 1 < steps) {
            async_copy_b128(&lds[(i + 1) & 1][loff], gsrc + (size_t)(i + 1) * 64);
            wait_async_le1();                          // oldest (current buf) done
        } else {
            wait_async_le0();
        }
        __syncthreads();                               // whole tile visible

        const unsigned char* bufc = lds[i & 1];
        Frag16 a, f0, f1, f2, f3;
#pragma unroll
        for (int v = 0; v < 8; ++v) {
            // A (16x32 bf16): k = half*16 + 2v
            int ka = (i << 5) + v * 2 + hf * 16;
            a.u[v] = Wp[arow + (ka >> 1)];
            // B (32x16 bf16): k = half*16 + 2v  -> 32 contiguous LDS bytes/lane
            unsigned int bo = (unsigned int)hf * 32 + v * 4;
            f0.u[v] = *(const unsigned int*)(bufc + (0 * 16 + lm) * LDS_PITCH + bo);
            f1.u[v] = *(const unsigned int*)(bufc + (1 * 16 + lm) * LDS_PITCH + bo);
            f2.u[v] = *(const unsigned int*)(bufc + (2 * 16 + lm) * LDS_PITCH + bo);
            f3.u[v] = *(const unsigned int*)(bufc + (3 * 16 + lm) * LDS_PITCH + bo);
        }
        acc0 = __builtin_amdgcn_wmma_f32_16x16x32_bf16(false, a.v, false, f0.v, (short)0, acc0, false, false);
        acc1 = __builtin_amdgcn_wmma_f32_16x16x32_bf16(false, a.v, false, f1.v, (short)0, acc1, false, false);
        acc2 = __builtin_amdgcn_wmma_f32_16x16x32_bf16(false, a.v, false, f2.v, (short)0, acc2, false, false);
        acc3 = __builtin_amdgcn_wmma_f32_16x16x32_bf16(false, a.v, false, f3.v, (short)0, acc3, false, false);

        __syncthreads();   // all waves done with buf before it is re-filled
    }

    // C/D layout: vgpr v -> m = v + 8*half; col n = lane%16
    float* Yb = Y + ((size_t)b * HCH + rt * 16) * NPTS + (size_t)ct * 64;
#pragma unroll
    for (int v = 0; v < 8; ++v) {
        int m = v + 8 * hf;
        Yb[(size_t)m * NPTS + lm +  0] = acc0[v];
        Yb[(size_t)m * NPTS + lm + 16] = acc1[v];
        Yb[(size_t)m * NPTS + lm + 32] = acc2[v];
        Yb[(size_t)m * NPTS + lm + 48] = acc3[v];
    }
}

// ---------------------------------------------------------------------------
// Kernel 4: per-channel BN stats over (B, N): stats[c] = {scale, shift}
// (conv bias cancels exactly under training-mode BN, so it is omitted)
// ---------------------------------------------------------------------------
__global__ void bn_stats(const float* __restrict__ Y, const float* __restrict__ g,
                         const float* __restrict__ beta, float* __restrict__ stats) {
    int c = blockIdx.x, tid = threadIdx.x;
    float s = 0.f, s2 = 0.f;
    for (int b = 0; b < BATCH; ++b) {
        const float* p = Y + ((size_t)b * HCH + c) * NPTS;
        for (int n = tid; n < NPTS; n += 256) {
            float v = p[n];
            s += v; s2 += v * v;
        }
    }
    __shared__ float sh[256], sh2[256];
    sh[tid] = s; sh2[tid] = s2;
    __syncthreads();
    for (int off = 128; off > 0; off >>= 1) {
        if (tid < off) { sh[tid] += sh[tid + off]; sh2[tid] += sh2[tid + off]; }
        __syncthreads();
    }
    if (tid == 0) {
        const float inv = 1.f / (float)(BATCH * NPTS);
        float mean = sh[0] * inv;
        float var  = sh2[0] * inv - mean * mean;
        float is   = rsqrtf(var + 1e-5f);
        float sc   = g[c] * is;
        stats[c * 2]     = sc;
        stats[c * 2 + 1] = beta[c] - mean * sc;
    }
}

// Kernel 5: BN+ReLU, f32 (B,H,N) -> bf16 X2 (B*N, H) for GEMM2
__global__ void bn_apply_bf16(const float* __restrict__ Y, const float* __restrict__ stats,
                              unsigned short* __restrict__ X2) {
    size_t i = (size_t)blockIdx.x * blockDim.x + threadIdx.x;
    int n = (int)(i % NPTS);
    int c = (int)((i / NPTS) % HCH);
    int b = (int)(i / ((size_t)NPTS * HCH));
    float v = fmaxf(Y[i] * stats[c * 2] + stats[c * 2 + 1], 0.f);
    X2[((size_t)b * NPTS + n) * HCH + c] = f32_to_bf16(v);
}

// Kernel 6: final BN+ReLU in place on d_out
__global__ void bn_apply_inplace(float* __restrict__ Y, const float* __restrict__ stats) {
    size_t i = (size_t)blockIdx.x * blockDim.x + threadIdx.x;
    int c = (int)((i / NPTS) % HCH);
    Y[i] = fmaxf(Y[i] * stats[c * 2] + stats[c * 2 + 1], 0.f);
}

// ---------------------------------------------------------------------------
extern "C" void kernel_launch(void* const* d_in, const int* in_sizes, int n_in,
                              void* d_out, int out_size, void* d_ws, size_t ws_size,
                              hipStream_t stream) {
    const float* xyz1    = (const float*)d_in[0];
    const float* xyz2    = (const float*)d_in[1];
    const float* points1 = (const float*)d_in[2];
    const float* points2 = (const float*)d_in[3];
    const float* W1      = (const float*)d_in[4];
    // d_in[5] = b1 (cancelled by BN), d_in[6]=g1, d_in[7]=beta1
    const float* g1      = (const float*)d_in[6];
    const float* beta1   = (const float*)d_in[7];
    const float* W2      = (const float*)d_in[8];
    const float* g2      = (const float*)d_in[10];
    const float* beta2   = (const float*)d_in[11];
    float* out = (float*)d_out;

    char* ws = (char*)d_ws;
    const size_t XB_BYTES  = (size_t)BATCH * NPTS * CIN * 2;  // 50,331,648 (also reused for X2)
    const size_t W1B_BYTES = (size_t)HCH * CIN * 2;
    const size_t W2B_BYTES = (size_t)HCH * HCH * 2;
    unsigned short* Xb     = (unsigned short*)ws;
    unsigned short* W1b    = (unsigned short*)(ws + XB_BYTES);
    unsigned short* W2b    = (unsigned short*)(ws + XB_BYTES + W1B_BYTES);
    float*          stats1 = (float*)(ws + XB_BYTES + W1B_BYTES + W2B_BYTES);
    float*          stats2 = stats1 + 2 * HCH;

    const int totalY   = BATCH * HCH * NPTS;                    // 16,777,216
    const int gemmBlks = BATCH * (HCH / 128) * (NPTS / 64);     // 2048 blocks x 256 thr

    cvt_weights<<<(HCH * CIN + 255) / 256, 256, 0, stream>>>(W1, W2, W1b, W2b);
    knn_interp<<<BATCH * (NPTS / 256), 256, 0, stream>>>(xyz1, xyz2, points1, points2, Xb);

    // Layer 1: GEMM (K=384) -> d_out (pre-BN), stats, BN+ReLU -> bf16 X2 (reuse Xb region)
    gemm_wmma_bf16<<<gemmBlks, 256, 0, stream>>>(W1b, Xb, out, CIN);
    bn_stats<<<HCH, 256, 0, stream>>>(out, g1, beta1, stats1);
    bn_apply_bf16<<<totalY / 256, 256, 0, stream>>>(out, stats1, Xb);

    // Layer 2: GEMM (K=256) -> d_out, stats, BN+ReLU in place
    gemm_wmma_bf16<<<gemmBlks, 256, 0, stream>>>(W2b, Xb, out, HCH);
    bn_stats<<<HCH, 256, 0, stream>>>(out, g2, beta2, stats2);
    bn_apply_inplace<<<totalY / 256, 256, 0, stream>>>(out, stats2);
}
